// LSTM_DNN_AC_4896262717928
// MI455X (gfx1250) — matrix operand order repaired and checked by reading, not compiled, not run on hardware
//
#include <hip/hip_runtime.h>

// ---------------------------------------------------------------------------
// LSTM_DNN_AC on MI455X (gfx1250, wave32, WMMA)
//
//  K1: xz1 = c_inp @ W1 + bL1  (WMMA bf16 GEMM; output stored in C-fragment
//      swizzled bf16 layout -> recurrent kernel initializes accumulators with
//      one global_load_b128 per lane, prefetched one step ahead)
//  K2: persistent recurrent kernel: 64 blocks x 512 threads (16 waves).
//      Each block owns 16 batch rows, loops t=0..511 over all 3 LSTM layers.
//      All recurrent weights staged in LDS as pre-swizzled WMMA B-fragments
//      (~236KB bf16; only possible with CDNA5's 320KB WGP LDS). h/c state
//      never leaves the block. Gates in f32.
//  K3: tiny dense heads, one thread per batch row.
// ---------------------------------------------------------------------------

typedef __attribute__((ext_vector_type(16))) __bf16 v16bf;
typedef __attribute__((ext_vector_type(8)))  __bf16 v8bf;
typedef __attribute__((ext_vector_type(8)))  float  v8f;

#define WMMA_BF16(a, b, c) \
  __builtin_amdgcn_wmma_f32_16x16x32_bf16(false, (a), false, (b), (short)0, (c), false, false)

__device__ __forceinline__ float sig_(float x) { return 1.0f / (1.0f + __expf(-x)); }
__device__ __forceinline__ float tanh_(float x) {
  // numerically safe: e->inf => 1, e->0 => -1
  float e = __expf(2.0f * x);
  return 1.0f - 2.0f / (e + 1.0f);
}

// LDS B-fragment read with a laundered 32-bit offset: prevents the compiler
// from proving the address loop-invariant (which caused hoist+scratch-spill),
// while keeping LDS provenance so it still lowers to ds_load_b128.
__device__ __forceinline__ v16bf ldsB(const __bf16* base, unsigned off_elems) {
  asm volatile("" : "+v"(off_elems));
  return *(const v16bf*)(base + off_elems);
}

// Stage a KxN f32 weight matrix into LDS as WMMA B-fragments (bf16).
// Fragment (nt, kt), per lane: n = lane&15, k = kt*32 + (lane>>4)*8 + (e&7) + (e>>3)*16.
// Storage: F[((nt*ktiles + kt)*32 + lane)*16 + e] -> per-lane contiguous v16bf.
__device__ __forceinline__ void stage_frags(__bf16* F, const float* Wg, int N, int Kreal,
                                            int ktiles, int ntiles, int tid, int nthreads) {
  int total = ntiles * ktiles * 512;
  for (int i = tid; i < total; i += nthreads) {
    int e = i & 15, ln = (i >> 4) & 31, tk = i >> 9;
    int kt = tk % ktiles, nt = tk / ktiles;
    int nn = ln & 15;
    int krow = kt * 32 + ((ln >> 4) << 3) + (e & 7) + ((e >> 3) << 4);
    F[i] = (krow < Kreal) ? (__bf16)Wg[krow * N + nt * 16 + nn] : (__bf16)0.0f;
  }
}

// Build a 16x32 bf16 A-fragment from an LDS row buffer with row stride `stride`
// (halves). ISA 7.12.2 pattern: half h=(lane>>4): elems 0..7 -> K=kbase+h*8+j,
// elems 8..15 -> K=kbase+h*8+16+j. Strides padded for conflict-free row reads.
__device__ __forceinline__ v16bf load_afrag(const __bf16* buf, int stride, int kbase, int lane) {
  int m = lane & 15, hi = lane >> 4;
  const __bf16* hp = buf + m * stride + kbase + hi * 8;
  v16bf a;
#pragma unroll
  for (int j = 0; j < 8; ++j) { a[j] = hp[j]; a[j + 8] = hp[16 + j]; }
  return a;
}

// ======================= K1: xz1 = c_inp @ W1 + bL1 ========================
// grid 2048 = 64 batch-tiles x 32 time-chunks(16 t each); 256 threads (8 waves).
__global__ void lstm1_xproj(const float* __restrict__ cinp, const float* __restrict__ W1,
                            const float* __restrict__ bL1, __bf16* __restrict__ xz1s) {
  extern __shared__ char smem[];
  __bf16* W1F = (__bf16*)smem;            // 32 ntiles * 2 ktiles * 512 halves = 64KB
  float*  b1s = (float*)(smem + 65536);   // 512 f32
  const int tid = threadIdx.x;
  const int lane = tid & 31, w = tid >> 5;
  const int bt = blockIdx.x >> 5;         // 0..63
  const int tc = blockIdx.x & 31;         // 0..31

  stage_frags(W1F, W1, 512, 64, 2, 32, tid, 256);
  for (int i = tid; i < 512; i += 256) b1s[i] = bL1[i];
  __syncthreads();

  const int m = lane & 15, hi = lane >> 4;
  // 16 t's x 8 ntile-groups(4 tiles) = 128 tasks over 8 waves
  for (int task = w; task < 128; task += 8) {
    int tt = task >> 3, ng = task & 7;
    int t = tc * 16 + tt;
    const float* arow = cinp + ((size_t)(bt * 16 + m) * 512 + t) * 64;
    v16bf afr[2];
#pragma unroll
    for (int kt = 0; kt < 2; ++kt) {
      int kb = kt * 32 + hi * 8;
#pragma unroll
      for (int j = 0; j < 8; ++j) {
        afr[kt][j]     = (__bf16)arow[kb + j];
        afr[kt][j + 8] = (__bf16)arow[kb + 16 + j];
      }
    }
#pragma unroll
    for (int q = 0; q < 4; ++q) {
      int nt = ng * 4 + q;
      float bv = b1s[nt * 16 + m];
      v8f acc = {bv, bv, bv, bv, bv, bv, bv, bv};
#pragma unroll
      for (int kt = 0; kt < 2; ++kt) {
        v16bf bfr = ldsB(W1F, (unsigned)(((nt * 2 + kt) * 32 + lane) * 16));
        acc = WMMA_BF16(afr[kt], bfr, acc);
      }
      v8bf outv;
#pragma unroll
      for (int r = 0; r < 8; ++r) outv[r] = (__bf16)acc[r];
      *(v8bf*)&xz1s[((((size_t)bt * 512 + t) * 32 + nt) * 32 + lane) * 8] = outv;
    }
  }
}

// ======================= K2: persistent 3-layer recurrence =================
// grid 64 (one block per 16 batch rows), 512 threads = 16 waves.
// Dynamic LDS layout (bytes):
//   U1F 0..131071 | W2F ..196607 | U2F ..229375 | W3F ..237567 | U3F ..241663
//   h1b(16x136 bf16) ..246015 | h2b(16x72) ..248319 | h3b(16x40) ..249599
//   zb(16x516 f32) ..282623 | b2s(256 f32) ..283647 | b3s(64 f32) ..283903
__global__ void lstm_recurrent(const __bf16* __restrict__ xz1s,
                               const float* __restrict__ U1, const float* __restrict__ W2,
                               const float* __restrict__ U2, const float* __restrict__ bL2,
                               const float* __restrict__ W3, const float* __restrict__ U3,
                               const float* __restrict__ bL3, float* __restrict__ h3out) {
  extern __shared__ char smem[];
  __bf16* U1F = (__bf16*)(smem);
  __bf16* W2F = (__bf16*)(smem + 131072);
  __bf16* U2F = (__bf16*)(smem + 196608);
  __bf16* W3F = (__bf16*)(smem + 229376);
  __bf16* U3F = (__bf16*)(smem + 237568);
  __bf16* h1b = (__bf16*)(smem + 241664);
  __bf16* h2b = (__bf16*)(smem + 246016);
  __bf16* h3b = (__bf16*)(smem + 248320);
  float*  zb  = (float*)(smem + 249600);
  float*  b2s = (float*)(smem + 282624);
  float*  b3s = (float*)(smem + 283648);

  const int tid = threadIdx.x;
  const int lane = tid & 31, w = tid >> 5;
  const int bt = blockIdx.x;
  const int m = lane & 15, hi = lane >> 4;

  stage_frags(U1F, U1, 512, 128, 4, 32, tid, 512);
  stage_frags(W2F, W2, 256, 128, 4, 16, tid, 512);
  stage_frags(U2F, U2, 256, 64, 2, 16, tid, 512);
  stage_frags(W3F, W3, 64, 64, 2, 4, tid, 512);
  stage_frags(U3F, U3, 64, 16, 1, 4, tid, 512);  // K padded 16->32 with zeros
  for (int i = tid; i < 16 * 136; i += 512) h1b[i] = (__bf16)0.0f;
  for (int i = tid; i < 16 * 72; i += 512)  h2b[i] = (__bf16)0.0f;
  for (int i = tid; i < 16 * 40; i += 512)  h3b[i] = (__bf16)0.0f;
  if (tid < 256) b2s[tid] = bL2[tid];
  if (tid < 64)  b3s[tid] = bL3[tid];
  __syncthreads();

  float c1r[4] = {0.f, 0.f, 0.f, 0.f};
  float c2r[2] = {0.f, 0.f};
  float c3r = 0.f;

  // prefetch xz for t=0; wave w owns layer-1 ntiles {2w, 2w+1}
  v8bf xz0, xz1;
  {
    const __bf16* p = xz1s + (((size_t)bt * 512 + 0) * 32 + w * 2) * 256;
    xz0 = *(const v8bf*)(p + lane * 8);
    xz1 = *(const v8bf*)(p + 256 + lane * 8);
  }

  for (int t = 0; t < 512; ++t) {
    // ---- layer 1: z1 = xz1_t + h1 @ U1 ----
    v16bf a1[4];
#pragma unroll
    for (int kt = 0; kt < 4; ++kt) a1[kt] = load_afrag(h1b, 136, kt * 32, lane);
    v8f acc0, acc1;
#pragma unroll
    for (int r = 0; r < 8; ++r) { acc0[r] = (float)xz0[r]; acc1[r] = (float)xz1[r]; }
    if (t + 1 < 512) {  // prefetch next step's xz (hidden behind WMMAs/gates)
      const __bf16* p = xz1s + (((size_t)bt * 512 + (t + 1)) * 32 + w * 2) * 256;
      xz0 = *(const v8bf*)(p + lane * 8);
      xz1 = *(const v8bf*)(p + 256 + lane * 8);
    }
#pragma unroll
    for (int kt = 0; kt < 4; ++kt) {
      v16bf b0 = ldsB(U1F, (unsigned)((((w * 2 + 0) * 4 + kt) * 32 + lane) * 16));
      acc0 = WMMA_BF16(a1[kt], b0, acc0);
      v16bf b1 = ldsB(U1F, (unsigned)((((w * 2 + 1) * 4 + kt) * 32 + lane) * 16));
      acc1 = WMMA_BF16(a1[kt], b1, acc1);
    }
#pragma unroll
    for (int r = 0; r < 8; ++r) {
      zb[(r + hi * 8) * 516 + (w * 2 + 0) * 16 + m] = acc0[r];
      zb[(r + hi * 8) * 516 + (w * 2 + 1) * 16 + m] = acc1[r];
    }
    __syncthreads();
    // gates L1: 16x128 elems, 4 per thread
#pragma unroll
    for (int u = 0; u < 4; ++u) {
      int idx = tid + u * 512;
      int mm = idx >> 7, j = idx & 127;
      const float* zr = &zb[mm * 516];
      float ii = sig_(zr[j]);
      float ff = sig_(zr[128 + j]);
      float gg = tanh_(zr[256 + j]);
      float oo = sig_(zr[384 + j]);
      float c = ff * c1r[u] + ii * gg;
      c1r[u] = c;
      h1b[mm * 136 + j] = (__bf16)(oo * tanh_(c));
    }
    __syncthreads();

    // ---- layer 2: z2 = h1 @ W2 + h2 @ U2 + b2 (wave w -> ntile w) ----
    {
      float bv = b2s[w * 16 + m];
      v8f acc = {bv, bv, bv, bv, bv, bv, bv, bv};
#pragma unroll
      for (int kt = 0; kt < 4; ++kt) {
        v16bf a = load_afrag(h1b, 136, kt * 32, lane);
        v16bf bb = ldsB(W2F, (unsigned)(((w * 4 + kt) * 32 + lane) * 16));
        acc = WMMA_BF16(a, bb, acc);
      }
#pragma unroll
      for (int kt = 0; kt < 2; ++kt) {
        v16bf a = load_afrag(h2b, 72, kt * 32, lane);
        v16bf bb = ldsB(U2F, (unsigned)(((w * 2 + kt) * 32 + lane) * 16));
        acc = WMMA_BF16(a, bb, acc);
      }
#pragma unroll
      for (int r = 0; r < 8; ++r) zb[(r + hi * 8) * 516 + w * 16 + m] = acc[r];
    }
    __syncthreads();
    // gates L2: 16x64 elems, 2 per thread
#pragma unroll
    for (int u = 0; u < 2; ++u) {
      int idx = tid + u * 512;
      int mm = idx >> 6, j = idx & 63;
      const float* zr = &zb[mm * 516];
      float ii = sig_(zr[j]);
      float ff = sig_(zr[64 + j]);
      float gg = tanh_(zr[128 + j]);
      float oo = sig_(zr[192 + j]);
      float c = ff * c2r[u] + ii * gg;
      c2r[u] = c;
      h2b[mm * 72 + j] = (__bf16)(oo * tanh_(c));
    }
    __syncthreads();

    // ---- layer 3: z3 = h2 @ W3 + h3 @ U3pad + b3 (waves 0..3) ----
    if (w < 4) {
      float bv = b3s[w * 16 + m];
      v8f acc = {bv, bv, bv, bv, bv, bv, bv, bv};
#pragma unroll
      for (int kt = 0; kt < 2; ++kt) {
        v16bf a = load_afrag(h2b, 72, kt * 32, lane);
        v16bf bb = ldsB(W3F, (unsigned)(((w * 2 + kt) * 32 + lane) * 16));
        acc = WMMA_BF16(a, bb, acc);
      }
      {
        v16bf a = load_afrag(h3b, 40, 0, lane);  // K 16..31 zero-padded
        v16bf bb = ldsB(U3F, (unsigned)((w * 32 + lane) * 16));
        acc = WMMA_BF16(a, bb, acc);
      }
#pragma unroll
      for (int r = 0; r < 8; ++r) zb[(r + hi * 8) * 516 + w * 16 + m] = acc[r];
    }
    __syncthreads();
    // gates L3: 16x16 elems, threads 0..255
    if (tid < 256) {
      int mm = tid >> 4, j = tid & 15;
      const float* zr = &zb[mm * 516];
      float ii = sig_(zr[j]);
      float ff = sig_(zr[16 + j]);
      float gg = tanh_(zr[32 + j]);
      float oo = sig_(zr[48 + j]);
      float c = ff * c3r + ii * gg;
      c3r = c;
      h3b[mm * 40 + j] = (__bf16)(oo * tanh_(c));
    }
    __syncthreads();
  }

  if (tid < 256) {
    int mm = tid >> 4, j = tid & 15;
    h3out[(bt * 16 + mm) * 16 + j] = (float)h3b[mm * 40 + j];
  }
}

// ======================= K3: dense heads ===================================
__global__ void head_kernel(const float* __restrict__ h3, const float* __restrict__ b_inp,
                            const float* __restrict__ Wd1, const float* __restrict__ bd1,
                            const float* __restrict__ Wd2, const float* __restrict__ bd2,
                            const float* __restrict__ Wsh, const float* __restrict__ bsh,
                            const float* __restrict__ Wp1, const float* __restrict__ bp1,
                            const float* __restrict__ Wp2, const float* __restrict__ bp2,
                            const float* __restrict__ Wv1, const float* __restrict__ bv1,
                            const float* __restrict__ Wv2, const float* __restrict__ bv2,
                            float* __restrict__ out) {
  int b = blockIdx.x * blockDim.x + threadIdx.x;
  if (b >= 1024) return;
  const float* bi = b_inp + b * 32;
  float v1[64];
#pragma unroll 4
  for (int j = 0; j < 64; ++j) {
    float s = bd1[j];
#pragma unroll
    for (int k = 0; k < 32; ++k) s += bi[k] * Wd1[k * 64 + j];
    v1[j] = fmaxf(s, 0.0f);
  }
  float x[32];
#pragma unroll
  for (int j = 0; j < 16; ++j) {
    float s = bd2[j];
#pragma unroll
    for (int k = 0; k < 64; ++k) s += v1[k] * Wd2[k * 16 + j];
    x[16 + j] = fmaxf(s, 0.0f);
  }
#pragma unroll
  for (int j = 0; j < 16; ++j) x[j] = h3[b * 16 + j];
  float accP[32], accV[32];
#pragma unroll
  for (int j = 0; j < 32; ++j) { accP[j] = bp1[j]; accV[j] = bv1[j]; }
  for (int jj = 0; jj < 128; ++jj) {
    float s = bsh[jj];
#pragma unroll
    for (int k = 0; k < 32; ++k) s += x[k] * Wsh[k * 128 + jj];
    s = fmaxf(s, 0.0f);
#pragma unroll
    for (int j = 0; j < 32; ++j) {
      accP[j] += s * Wp1[jj * 32 + j];
      accV[j] += s * Wv1[jj * 32 + j];
    }
  }
  float p0 = bp2[0], p1 = bp2[1], p2 = bp2[2], val = bv2[0];
#pragma unroll
  for (int j = 0; j < 32; ++j) {
    float rp = fmaxf(accP[j], 0.0f);
    float rv = fmaxf(accV[j], 0.0f);
    p0 += rp * Wp2[j * 3 + 0];
    p1 += rp * Wp2[j * 3 + 1];
    p2 += rp * Wp2[j * 3 + 2];
    val += rv * Wv2[j];
  }
  out[b * 3 + 0] = p0;
  out[b * 3 + 1] = p1;
  out[b * 3 + 2] = p2;
  out[3 * 1024 + b] = val;
}

extern "C" void kernel_launch(void* const* d_in, const int* in_sizes, int n_in,
                              void* d_out, int out_size, void* d_ws, size_t ws_size,
                              hipStream_t stream) {
  (void)in_sizes; (void)n_in; (void)out_size; (void)ws_size;
  const float* c_inp = (const float*)d_in[0];
  const float* b_inp = (const float*)d_in[1];
  const float* W1  = (const float*)d_in[2];
  const float* U1  = (const float*)d_in[3];
  const float* bL1 = (const float*)d_in[4];
  const float* W2  = (const float*)d_in[5];
  const float* U2  = (const float*)d_in[6];
  const float* bL2 = (const float*)d_in[7];
  const float* W3  = (const float*)d_in[8];
  const float* U3  = (const float*)d_in[9];
  const float* bL3 = (const float*)d_in[10];
  const float* Wd1 = (const float*)d_in[11];
  const float* bd1 = (const float*)d_in[12];
  const float* Wd2 = (const float*)d_in[13];
  const float* bd2 = (const float*)d_in[14];
  const float* Wsh = (const float*)d_in[15];
  const float* bsh = (const float*)d_in[16];
  const float* Wp1 = (const float*)d_in[17];
  const float* bp1 = (const float*)d_in[18];
  const float* Wp2 = (const float*)d_in[19];
  const float* bp2 = (const float*)d_in[20];
  const float* Wv1 = (const float*)d_in[21];
  const float* bv1 = (const float*)d_in[22];
  const float* Wv2 = (const float*)d_in[23];
  const float* bv2 = (const float*)d_in[24];

  __bf16* xz1s = (__bf16*)d_ws;                              // 512 MiB swizzled xz1
  float*  h3g  = (float*)((char*)d_ws + (size_t)536870912);  // [1024,16] f32

  lstm1_xproj<<<2048, 256, 67584, stream>>>(c_inp, W1, bL1, xz1s);
  lstm_recurrent<<<64, 512, 283904, stream>>>(xz1s, U1, W2, U2, bL2, W3, U3, bL3, h3g);
  head_kernel<<<4, 256, 0, stream>>>(h3g, b_inp, Wd1, bd1, Wd2, bd2, Wsh, bsh,
                                     Wp1, bp1, Wp2, bp2, Wv1, bv1, Wv2, bv2,
                                     (float*)d_out);
}
